// CAMMambaBlock_8641474199632
// MI455X (gfx1250) — compile-verified, matrix operand order checked by reading, not run on hardware
//
#include <hip/hip_runtime.h>
#include <hip/hip_bf16.h>
#include <stdint.h>

// CAM-Mamba block for MI455X (gfx1250, wave32).
// Pipeline: [pad x_proj_w] -> [LN + in_proj WMMA] -> [causal depthwise conv + SiLU]
//           -> [x_proj WMMA + dt_proj + softplus] -> [chunked selective scan (3 kernels)]
//           -> [out_proj WMMA + residual].
// fp32 WMMA 16x16x4 (confirmed round 1). 4 M-tiles share each B fragment -> 4x WMMA ILP.
// Activation tiles staged into LDS via the Tensor Data Mover (strided 2D tile +
// hardware LDS padding), synchronized with s_wait_tensorcnt.

typedef float    v2f  __attribute__((ext_vector_type(2)));
typedef float    v8f  __attribute__((ext_vector_type(8)));
typedef _Float16 v16h __attribute__((ext_vector_type(16)));
typedef unsigned int v4u __attribute__((ext_vector_type(4)));
typedef int      v4i  __attribute__((ext_vector_type(4)));
typedef int      v8i  __attribute__((ext_vector_type(8)));

#if defined(__has_builtin)
#  if __has_builtin(__builtin_amdgcn_wmma_f32_16x16x4_f32)
#    define WMMA_F32_NATIVE 1
#  endif
#endif
#ifndef WMMA_F32_NATIVE
#  define WMMA_F32_NATIVE 0
#endif

#if defined(__AMDGCN__) && defined(__has_builtin)
#  if __has_builtin(__builtin_amdgcn_tensor_load_to_lds) && \
      __has_builtin(__builtin_amdgcn_s_wait_tensorcnt)
#    define HAVE_TDM 1
#  endif
#endif
#ifndef HAVE_TDM
#  define HAVE_TDM 0
#endif

#define DIMC   128
#define LVAL   9216          // H*W
#define NB     8
#define NCHUNK 36
#define CHUNK  256           // 36*256 == 9216

#if HAVE_TDM
// Issue one TDM 2D tile load: dim0 contiguous elements per row (f32), `rows` rows,
// global row stride `stride0` elements, with LDS padding of (pad_amt code) DWORDs
// every (pad_int code) DWORDs.  Descriptor layout per CDNA5 ISA.
__device__ inline void tdm_load_2d(unsigned lds_addr, const void* gsrc,
                                   unsigned dim0, unsigned rows, unsigned stride0,
                                   unsigned pad_int, unsigned pad_amt) {
  uint64_t ga = (uint64_t)(uintptr_t)gsrc;
  v4u g0;
  g0.x = 1u;                                                 // count=1 user D#
  g0.y = lds_addr;                                           // LDS byte address
  g0.z = (unsigned)ga;                                       // global_addr[31:0]
  g0.w = (unsigned)((ga >> 32) & 0x01FFFFFFu) | (2u << 30);  // [56:32] | type=2
  v8i g1;
  g1[0] = (int)((2u << 16) | (1u << 20) | (pad_int << 22) | (pad_amt << 25));
  g1[1] = (int)(dim0 << 16);                     // tensor_dim0[15:0]  (bits 63:48)
  g1[2] = (int)((dim0 >> 16) | (rows << 16));    // dim0 hi | tensor_dim1[15:0]
  g1[3] = (int)((rows >> 16) | (dim0 << 16));    // dim1 hi | tile_dim0 (== dim0)
  g1[4] = (int)rows;                             // tile_dim1 (tile_dim2 = 0)
  g1[5] = (int)stride0;                          // tensor_dim0_stride lo32
  g1[6] = 0;
  g1[7] = 0;
  v4i g2 = {0, 0, 0, 0};
  v4i g3 = {0, 0, 0, 0};
#if defined(__clang_major__) && (__clang_major__ >= 23)
  v8i g4 = {0, 0, 0, 0, 0, 0, 0, 0};
  __builtin_amdgcn_tensor_load_to_lds(g0, g1, g2, g3, g4, 0);
#else
  __builtin_amdgcn_tensor_load_to_lds(g0, g1, g2, g3, 0);
#endif
}
#endif

// Compute a 64x16 strip: 4 M-tiles x one N-tile.  A element (m,k) at A[m*am + k*ak]
// (LDS, row- or K-major).  B = 16 rows of an (N x 128) weight matrix (global).
// One B fragment feeds 4 independent WMMA accumulator chains.
__device__ inline void wmma_strip_k128(const float* A, int am, int ak,
                                       const float* __restrict__ B, v8f acc[4]) {
  const int lane = threadIdx.x & 31;
  const int r = lane & 15, hf = lane >> 4;
#if WMMA_F32_NATIVE
  for (int k0 = 0; k0 < 128; k0 += 4) {
    v2f bfr;
    bfr.x = B[r * 128 + k0 + 2 * hf];
    bfr.y = B[r * 128 + k0 + 2 * hf + 1];
#pragma unroll
    for (int mt = 0; mt < 4; ++mt) {
      v2f afr;
      afr.x = A[(mt * 16 + r) * am + (k0 + 2 * hf) * ak];
      afr.y = A[(mt * 16 + r) * am + (k0 + 2 * hf + 1) * ak];
      acc[mt] = __builtin_amdgcn_wmma_f32_16x16x4_f32(false, afr, false, bfr,
                                                      (short)0, acc[mt], false, false);
    }
  }
#else
  for (int k0 = 0; k0 < 128; k0 += 32) {
    v16h bfr;
#pragma unroll
    for (int j = 0; j < 8; ++j) {
      bfr[j]     = (_Float16)B[r * 128 + k0 + 8 * hf + j];
      bfr[8 + j] = (_Float16)B[r * 128 + k0 + 16 + 8 * hf + j];
    }
#pragma unroll
    for (int mt = 0; mt < 4; ++mt) {
      v16h afr;
#pragma unroll
      for (int j = 0; j < 8; ++j) {
        afr[j]     = (_Float16)A[(mt * 16 + r) * am + (k0 + 8 * hf + j) * ak];
        afr[8 + j] = (_Float16)A[(mt * 16 + r) * am + (k0 + 16 + 8 * hf + j) * ak];
      }
      acc[mt] = __builtin_amdgcn_wmma_f32_16x16x32_f16(false, afr, false, bfr,
                                                       (short)0, acc[mt], false, false);
    }
  }
#endif
}

__device__ inline v8f zero8() {
  v8f z;
#pragma unroll
  for (int e = 0; e < 8; ++e) z[e] = 0.0f;
  return z;
}

// ---------------- K0: zero-pad x_proj_w from 40x128 to 48x128 -------------------
__global__ void k_pad_xw(const float* __restrict__ xw, float* __restrict__ xwpad) {
  int i = blockIdx.x * 256 + threadIdx.x;
  if (i < 48 * 128) {
    int j = i >> 7;
    xwpad[i] = (j < 40) ? xw[i] : 0.0f;
  }
}

// ---------------- K1: LayerNorm + in_proj (GEMM 64x128x256) ---------------------
// Tile held K-major in LDS: Xt[c][l], stride 66 -> natural for the TDM strided
// 2D gather (row = 64 contiguous l, 128 rows with stride 9216) and conflict-free
// WMMA A-fragment reads.
__global__ void __launch_bounds__(256)
k_ln_inproj(const float* __restrict__ inp, const float* __restrict__ lnw,
            const float* __restrict__ lnb, const float* __restrict__ Win,
            float* __restrict__ xm, float* __restrict__ z) {
  __shared__ float Xt[128 * 66];
  __shared__ float s_mu[64], s_rs[64];
  const int b = blockIdx.y, l0 = blockIdx.x * 64, tid = threadIdx.x;
#if HAVE_TDM
  if (tid < 32) {                       // wave 0 issues the DMA (EXEC-independent)
    tdm_load_2d((unsigned)(uintptr_t)&Xt[0],
                inp + ((size_t)b * 128) * LVAL + l0,
                /*dim0=*/64, /*rows=*/128, /*stride0=*/LVAL,
                /*pad: +2 dw per 64 dw*/ 5u, 1u);
    __builtin_amdgcn_s_wait_tensorcnt(0);
  }
  __syncthreads();
#else
  for (int idx = tid; idx < 64 * 128; idx += 256) {
    int c = idx >> 6, l = idx & 63;
    Xt[c * 66 + l] = inp[((size_t)b * 128 + c) * LVAL + l0 + l];
  }
  __syncthreads();
#endif
  if (tid < 64) {
    float s = 0.f, s2 = 0.f;
    for (int c = 0; c < 128; ++c) { float v = Xt[c * 66 + tid]; s += v; s2 += v * v; }
    float mu  = s * (1.0f / 128.0f);
    float var = s2 * (1.0f / 128.0f) - mu * mu;
    s_mu[tid] = mu;
    s_rs[tid] = rsqrtf(var + 1e-5f);
  }
  __syncthreads();
  for (int idx = tid; idx < 64 * 128; idx += 256) {
    int c = idx >> 6, l = idx & 63;
    Xt[c * 66 + l] = (Xt[c * 66 + l] - s_mu[l]) * s_rs[l] * lnw[c] + lnb[c];
  }
  __syncthreads();

  const int w = tid >> 5, lane = tid & 31;
  const int r = lane & 15, hf = lane >> 4;
  for (int halfsel = 0; halfsel < 2; ++halfsel) {    // strip 1 -> xm, strip 2 -> z
    const int n0 = halfsel * 128 + w * 16;
    v8f acc[4];
#pragma unroll
    for (int i = 0; i < 4; ++i) acc[i] = zero8();
    wmma_strip_k128(Xt, /*am=*/1, /*ak=*/66, Win + (size_t)n0 * 128, acc);
    float* dst = halfsel ? z : xm;
    const int cc = w * 16 + r;
#pragma unroll
    for (int mt = 0; mt < 4; ++mt) {
      size_t pb = (size_t)b * LVAL + l0 + mt * 16 + 8 * hf;
#pragma unroll
      for (int r2 = 0; r2 < 8; ++r2)
        dst[(pb + r2) * 128 + cc] = acc[mt][r2];
    }
  }
}

// ---------------- K2: depthwise causal conv1d + bias + SiLU ---------------------
__global__ void __launch_bounds__(256)
k_conv(const float* __restrict__ xm, const float* __restrict__ cw,
       const float* __restrict__ cb, float* __restrict__ xs) {
  int g = blockIdx.x * 256 + threadIdx.x;      // over BL*128
  int c = g & 127;
  size_t p = (size_t)(g >> 7);
  int l = (int)(p % LVAL);                     // causal pad is per-batch
  float v = cb[c];
#pragma unroll
  for (int k = 0; k < 4; ++k) {
    int dl = 3 - k;
    if (l >= dl) v += cw[c * 4 + k] * xm[(p - dl) * 128 + c];
  }
  float sig = 1.0f / (1.0f + __expf(-v));
  xs[g] = v * sig;
}

// ---------------- K3: x_proj (GEMM 64x128x48) + dt_proj + softplus --------------
__global__ void __launch_bounds__(256)
k_xproj(const float* __restrict__ xs, const float* __restrict__ xwpad,
        const float* __restrict__ dtw, const float* __restrict__ dtb,
        float* __restrict__ dtf, float* __restrict__ Bm, float* __restrict__ Cm) {
  __shared__ float X[64 * 132];
  __shared__ float DBL[64 * 52];
  const int b = blockIdx.y, l0 = blockIdx.x * 64, tid = threadIdx.x;
  const size_t p0 = (size_t)b * LVAL + l0;
#if HAVE_TDM
  if (tid < 32) {
    tdm_load_2d((unsigned)(uintptr_t)&X[0], xs + p0 * 128,
                /*dim0=*/128, /*rows=*/64, /*stride0=*/128,
                /*pad: +4 dw per 128 dw*/ 6u, 3u);
    __builtin_amdgcn_s_wait_tensorcnt(0);
  }
  __syncthreads();
#else
  for (int idx = tid; idx < 64 * 128; idx += 256) {
    int l = idx >> 7, c = idx & 127;
    X[l * 132 + c] = xs[(p0 + l) * 128 + c];
  }
  __syncthreads();
#endif
  const int w = tid >> 5, lane = tid & 31;
  const int r = lane & 15, hf = lane >> 4;
  if (w < 3) {
    const int n0 = w * 16;
    v8f acc[4];
#pragma unroll
    for (int i = 0; i < 4; ++i) acc[i] = zero8();
    wmma_strip_k128(X, /*am=*/132, /*ak=*/1, xwpad + (size_t)n0 * 128, acc);
#pragma unroll
    for (int mt = 0; mt < 4; ++mt)
#pragma unroll
      for (int r2 = 0; r2 < 8; ++r2)
        DBL[(mt * 16 + r2 + 8 * hf) * 52 + n0 + r] = acc[mt][r2];
  }
  __syncthreads();
  for (int idx = tid; idx < 64 * 128; idx += 256) {
    int l = idx >> 7, c = idx & 127;
    float s = dtb[c];
#pragma unroll
    for (int i = 0; i < 8; ++i) s += DBL[l * 52 + i] * dtw[c * 8 + i];
    float sp = (s > 20.0f) ? s : log1pf(__expf(s));   // softplus
    dtf[(p0 + l) * 128 + c] = sp;
  }
  for (int idx = tid; idx < 64 * 32; idx += 256) {
    int l = idx >> 5, j = idx & 31;
    if (j < 16) Bm[(p0 + l) * 16 + j]        = DBL[l * 52 + 8 + j];
    else        Cm[(p0 + l) * 16 + (j - 16)] = DBL[l * 52 + 24 + (j - 16)];
  }
}

// ---------------- K4: per-chunk local scan + chunk aggregates -------------------
// thread t owns channel c=t>>1, states n0..n0+7 with n0=(t&1)*8  (2048 pairs/batch)
__global__ void __launch_bounds__(256)
k_scan_local(const float* __restrict__ dtf, const float* __restrict__ xs,
             const float* __restrict__ Bm, const float* __restrict__ Alog,
             float* __restrict__ chA, float* __restrict__ chH) {
  __shared__ float SB[16 * 16];
  const int b = blockIdx.y, ch = blockIdx.x, t = threadIdx.x;
  const int c = t >> 1, n0 = (t & 1) * 8;
  float An[8], ap[8], h[8];
#pragma unroll
  for (int j = 0; j < 8; ++j) {
    An[j] = -__expf(Alog[c * 16 + n0 + j]);
    ap[j] = 1.0f; h[j] = 0.0f;
  }
  const size_t pbase = (size_t)b * LVAL + ch * CHUNK;
  for (int g = 0; g < CHUNK; g += 16) {
    __syncthreads();
    { int s = t >> 4, n = t & 15; SB[t] = Bm[(pbase + g + s) * 16 + n]; }
    __syncthreads();
    if (g + 16 < CHUNK)
      __builtin_prefetch(&dtf[(pbase + g + 16) * 128 + c], 0, 1);
    for (int s = 0; s < 16; ++s) {
      size_t p = pbase + g + s;
      float dtv = dtf[p * 128 + c];
      float dx  = dtv * xs[p * 128 + c];
#pragma unroll
      for (int j = 0; j < 8; ++j) {
        float dA = __expf(dtv * An[j]);
        h[j]  = h[j] * dA + dx * SB[s * 16 + n0 + j];
        ap[j] *= dA;
      }
    }
  }
  size_t o = ((size_t)(b * NCHUNK + ch)) * 2048 + t * 8;
#pragma unroll
  for (int j = 0; j < 8; ++j) { chA[o + j] = ap[j]; chH[o + j] = h[j]; }
}

// ---------------- K5: sequential prefix over 36 chunk aggregates ----------------
__global__ void __launch_bounds__(256)
k_scan_prefix(const float* __restrict__ chA, const float* __restrict__ chH,
              float* __restrict__ chI) {
  const int b = blockIdx.x, t = threadIdx.x;
  float h[8];
#pragma unroll
  for (int j = 0; j < 8; ++j) h[j] = 0.0f;
  for (int ch = 0; ch < NCHUNK; ++ch) {
    size_t o = ((size_t)(b * NCHUNK + ch)) * 2048 + t * 8;
#pragma unroll
    for (int j = 0; j < 8; ++j) {
      chI[o + j] = h[j];
      h[j] = chA[o + j] * h[j] + chH[o + j];
    }
  }
}

// ---------------- K6: replay scan with prefix; fuse D-skip + SiLU(z) gate -------
__global__ void __launch_bounds__(256)
k_scan_out(const float* __restrict__ dtf, const float* __restrict__ xs,
           const float* __restrict__ Bm, const float* __restrict__ Cm,
           const float* __restrict__ z, const float* __restrict__ Alog,
           const float* __restrict__ Dv, const float* __restrict__ chI,
           float* __restrict__ yg) {
  __shared__ float SB[16 * 16], SC[16 * 16];
  const int b = blockIdx.y, ch = blockIdx.x, t = threadIdx.x;
  const int c = t >> 1, n0 = (t & 1) * 8;
  float An[8], h[8];
  const size_t o = ((size_t)(b * NCHUNK + ch)) * 2048 + t * 8;
#pragma unroll
  for (int j = 0; j < 8; ++j) {
    An[j] = -__expf(Alog[c * 16 + n0 + j]);
    h[j] = chI[o + j];
  }
  const float Dc = Dv[c];
  const size_t pbase = (size_t)b * LVAL + ch * CHUNK;
  for (int g = 0; g < CHUNK; g += 16) {
    __syncthreads();
    { int s = t >> 4, n = t & 15;
      SB[t] = Bm[(pbase + g + s) * 16 + n];
      SC[t] = Cm[(pbase + g + s) * 16 + n]; }
    __syncthreads();
    if (g + 16 < CHUNK)
      __builtin_prefetch(&dtf[(pbase + g + 16) * 128 + c], 0, 1);
    for (int s = 0; s < 16; ++s) {
      size_t p = pbase + g + s;
      float dtv = dtf[p * 128 + c];
      float xv  = xs[p * 128 + c];
      float dx  = dtv * xv;
      float yp  = 0.0f;
#pragma unroll
      for (int j = 0; j < 8; ++j) {
        float dA = __expf(dtv * An[j]);
        h[j] = h[j] * dA + dx * SB[s * 16 + n0 + j];
        yp  += h[j] * SC[s * 16 + n0 + j];
      }
      yp += __shfl_xor(yp, 1, 32);       // combine the two n-halves (same wave)
      if ((t & 1) == 0) {
        float yfull = yp + xv * Dc;
        float zv  = z[p * 128 + c];
        float sig = 1.0f / (1.0f + __expf(-zv));
        yg[p * 128 + c] = yfull * (zv * sig);
      }
    }
  }
}

// ---------------- K7: out_proj (GEMM 64x128x128) + residual ---------------------
__global__ void __launch_bounds__(256)
k_outproj(const float* __restrict__ yg, const float* __restrict__ Wout,
          const float* __restrict__ inp, float* __restrict__ out) {
  __shared__ float S[128 * 68];          // reused: 64x132 input tile then 128x68 out^T
  const int b = blockIdx.y, l0 = blockIdx.x * 64, tid = threadIdx.x;
  const size_t p0 = (size_t)b * LVAL + l0;
#if HAVE_TDM
  if (tid < 32) {
    tdm_load_2d((unsigned)(uintptr_t)&S[0], yg + p0 * 128,
                /*dim0=*/128, /*rows=*/64, /*stride0=*/128, 6u, 3u);
    __builtin_amdgcn_s_wait_tensorcnt(0);
  }
  __syncthreads();
#else
  for (int idx = tid; idx < 64 * 128; idx += 256) {
    int l = idx >> 7, c = idx & 127;
    S[l * 132 + c] = yg[(p0 + l) * 128 + c];
  }
  __syncthreads();
#endif
  const int w = tid >> 5, lane = tid & 31;
  const int r = lane & 15, hf = lane >> 4;
  const int n0 = w * 16;                 // one N-tile per wave, all 4 M-tiles
  v8f acc[4];
#pragma unroll
  for (int i = 0; i < 4; ++i) acc[i] = zero8();
  wmma_strip_k128(S, /*am=*/132, /*ak=*/1, Wout + (size_t)n0 * 128, acc);
  __syncthreads();                       // input tile dead; reuse LDS transposed
  const int cc = n0 + r;
#pragma unroll
  for (int mt = 0; mt < 4; ++mt)
#pragma unroll
    for (int r2 = 0; r2 < 8; ++r2)
      S[cc * 68 + (mt * 16 + r2 + 8 * hf)] = acc[mt][r2];
  __syncthreads();
  for (int idx = tid; idx < 64 * 128; idx += 256) {
    int c = idx >> 6, l = idx & 63;      // coalesced over l in (b,c,h,w) layout
    size_t gi = ((size_t)b * 128 + c) * LVAL + l0 + l;
    out[gi] = S[c * 68 + l] + inp[gi];
  }
}

// ------------------------------- launcher ---------------------------------------
extern "C" void kernel_launch(void* const* d_in, const int* in_sizes, int n_in,
                              void* d_out, int out_size, void* d_ws, size_t ws_size,
                              hipStream_t stream) {
  (void)in_sizes; (void)n_in; (void)out_size; (void)ws_size;
  const float* inp  = (const float*)d_in[0];
  const float* lnw  = (const float*)d_in[1];
  const float* lnb  = (const float*)d_in[2];
  const float* Win  = (const float*)d_in[3];
  const float* cw   = (const float*)d_in[4];
  const float* cb   = (const float*)d_in[5];
  const float* xpw  = (const float*)d_in[6];
  const float* dtw  = (const float*)d_in[7];
  const float* dtb  = (const float*)d_in[8];
  const float* Alog = (const float*)d_in[9];
  const float* Dv   = (const float*)d_in[10];
  const float* Wout = (const float*)d_in[11];
  float* out = (float*)d_out;

  const size_t BL = (size_t)NB * LVAL;
  float* ws  = (float*)d_ws;
  float* xm  = ws;                      // BL*128   (reused as yg after conv)
  float* zb  = xm  + BL * 128;          // BL*128
  float* xs  = zb  + BL * 128;          // BL*128
  float* dtf = xs  + BL * 128;          // BL*128
  float* Bm  = dtf + BL * 128;          // BL*16
  float* Cm  = Bm  + BL * 16;           // BL*16
  float* xwp = Cm  + BL * 16;           // 48*128
  float* chA = xwp + 48 * 128;          // NB*36*2048
  float* chH = chA + (size_t)NB * NCHUNK * 2048;
  float* chI = chH + (size_t)NB * NCHUNK * 2048;

  dim3 blk(256);
  k_pad_xw   <<<24, blk, 0, stream>>>(xpw, xwp);
  k_ln_inproj<<<dim3(LVAL / 64, NB), blk, 0, stream>>>(inp, lnw, lnb, Win, xm, zb);
  k_conv     <<<(unsigned)((BL * 128) / 256), blk, 0, stream>>>(xm, cw, cb, xs);
  k_xproj    <<<dim3(LVAL / 64, NB), blk, 0, stream>>>(xs, xwp, dtw, dtb, dtf, Bm, Cm);
  k_scan_local <<<dim3(NCHUNK, NB), blk, 0, stream>>>(dtf, xs, Bm, Alog, chA, chH);
  k_scan_prefix<<<NB, blk, 0, stream>>>(chA, chH, chI);
  k_scan_out <<<dim3(NCHUNK, NB), blk, 0, stream>>>(dtf, xs, Bm, Cm, zb, Alog, Dv,
                                                    chI, /*yg=*/xm);
  k_outproj  <<<dim3(LVAL / 64, NB), blk, 0, stream>>>(/*yg=*/xm, Wout, inp, out);
}
